// SyncConv_50019189129826
// MI455X (gfx1250) — matrix operand order repaired and checked by reading, not compiled, not run on hardware
//
#include <hip/hip_runtime.h>
#include <hip/hip_bf16.h>

#define B_      2
#define NV_     15000
#define NDIR    16
#define NRING   3
#define NCH     32
#define NF_     64
#define TV      8                   // vertices per block -> M = 16 rows (v,b)
#define NCHUNK  (NRING * NDIR)      // 48 K-chunks of 32 (one neighbor (r,d) each)
#define NBUF    3                   // LDS A-tile buffers (async pipeline depth)

// LDS A-tile strides (bf16 elems), padded for bank stagger
#define ROW_STRIDE 40               // 32 ch + 8 pad  (80 B, 16B aligned)
#define L_STRIDE   (16 * ROW_STRIDE + 8)   // 648 elems (1296 B, 16B aligned)

#define YBF_ELEMS  ((size_t)B_ * NV_ * NDIR * NCH)          // 15,360,000
#define WFRAG_OFF  (YBF_ELEMS * 2)                          // bytes into d_ws

typedef __attribute__((ext_vector_type(16))) __bf16 bf16x16;
typedef __attribute__((ext_vector_type(8)))  __bf16 bf16x8;
typedef __attribute__((ext_vector_type(8)))  float  floatx8;

// ---------------------------------------------------------------------------
// Prep 1: convert y (f32) -> ybf (bf16), same linear layout.
// ---------------------------------------------------------------------------
__global__ __launch_bounds__(256) void yprep_kernel(const float* __restrict__ y,
                                                    __bf16* __restrict__ ybf) {
    size_t i = ((size_t)blockIdx.x * 256 + threadIdx.x) * 8;   // 8 elems/thread
    const float4* src = (const float4*)(y + i);
    float4 a = src[0], b = src[1];
    union { __bf16 h[8]; uint4 u; } pk;
    pk.h[0] = (__bf16)a.x; pk.h[1] = (__bf16)a.y;
    pk.h[2] = (__bf16)a.z; pk.h[3] = (__bf16)a.w;
    pk.h[4] = (__bf16)b.x; pk.h[5] = (__bf16)b.y;
    pk.h[6] = (__bf16)b.z; pk.h[7] = (__bf16)b.w;
    *(uint4*)(ybf + i) = pk.u;
}

// ---------------------------------------------------------------------------
// Prep 2: pack kernel[r,(d-l)%16,c,f] into bf16 WMMA-B fragments.
// Layout: wfrag[l][kc=(r,d)][nt][lane][16 contiguous bf16]
//   lane < 16 : column n = lane,      K(=c) = 0..15
//   lane >=16 : column n = lane-16,   K(=c) = 16..31
// ---------------------------------------------------------------------------
__global__ __launch_bounds__(256) void wprep_kernel(const float* __restrict__ kern,
                                                    __bf16* __restrict__ wfrag) {
    int idx  = blockIdx.x * 256 + threadIdx.x;      // < 16*48*4*32*16 = 1,572,864
    int j    = idx & 15;
    int lane = (idx >> 4) & 31;
    int nt   = (idx >> 9) & 3;
    int lkc  = idx >> 11;                           // l*48 + kc
    int kc   = lkc % NCHUNK;
    int l    = lkc / NCHUNK;
    int r    = kc >> 4, d = kc & 15;
    int dd   = (d - l) & 15;
    int c    = ((lane & 16) ? 16 : 0) + j;
    int f    = nt * 16 + (lane & 15);
    float v  = kern[(((r * NDIR + dd) * NCH) + c) * NF_ + f];
    wfrag[idx] = (__bf16)v;
}

// ---------------------------------------------------------------------------
// Main conv: per block, TV vertices; per wave, 2 output directions.
// Gather runs on the ASYNCcnt path (global_load_async_to_lds_b128) into a
// triple-buffered LDS A-tile set; one raw split-barrier per chunk, no
// LOADcnt/DScnt drain at the barrier.
// ---------------------------------------------------------------------------
__global__ __launch_bounds__(256) void conv_kernel(const __bf16* __restrict__ ybf,
                                                   const int*   __restrict__ ftv,
                                                   const int*   __restrict__ ftd,
                                                   const float* __restrict__ bias,
                                                   const __bf16* __restrict__ wfrag,
                                                   float* __restrict__ out) {
    __shared__ __bf16 sA[NBUF][16 * L_STRIDE];      // rotating per-l A tiles
    __shared__ int    sV[TV * NCHUNK];
    __shared__ int    sD[TV * NCHUNK];

    const int tid  = threadIdx.x;
    const int lane = tid & 31;
    const int wave = tid >> 5;
    const int v0   = blockIdx.x * TV;

    // preload per-vertex neighbor indices (contiguous per vertex)
    if (tid < TV * NCHUNK) {
        int vloc = tid / NCHUNK, k = tid % NCHUNK;
        sV[tid] = ftv[(v0 + vloc) * NCHUNK + k];
        sD[tid] = ftd[(v0 + vloc) * NCHUNK + k];
    }

    // staging role: thread = (vertex, b, delta); gathers one 64B bf16 row
    const int svloc = tid >> 5;
    const int sb    = (tid >> 4) & 1;
    const int sdel  = tid & 15;
    const int srow  = svloc * 2 + sb;

    // compute role: wave owns directions l0, l0+1
    const int l0 = wave * 2;
    floatx8 acc[2][4];
    const floatx8 zf8 = {0.f, 0.f, 0.f, 0.f, 0.f, 0.f, 0.f, 0.f};
#pragma unroll
    for (int li = 0; li < 2; ++li)
#pragma unroll
        for (int nt = 0; nt < 4; ++nt) acc[li][nt] = zf8;

    float bval[4];
#pragma unroll
    for (int nt = 0; nt < 4; ++nt) bval[nt] = bias[nt * 16 + (lane & 15)];

    __syncthreads();                                // sV/sD ready (one full barrier)

    // async gather of chunk kc into buffer nb:
    //   4x global_load_async_to_lds_b128, ASYNCcnt-tracked; the VDST VGPR is
    //   the wave-relative LDS byte offset (low 32 bits of the generic pointer)
#define ISSUE_GATHER(KC, NB)                                                     \
    do {                                                                         \
        int nv_ = sV[svloc * NCHUNK + (KC)];                                     \
        int nd_ = sD[svloc * NCHUNK + (KC)];                                     \
        int l_  = (sdel - nd_) & 15;                                             \
        unsigned lds_ = (unsigned)(size_t)&sA[(NB)][l_ * L_STRIDE +              \
                                               srow * ROW_STRIDE];               \
        const __bf16* g_ =                                                       \
            ybf + (((size_t)sb * NV_ + nv_) * NDIR + sdel) * NCH;                \
        asm volatile(                                                            \
            "global_load_async_to_lds_b128 %0, %1, off\n\t"                      \
            "global_load_async_to_lds_b128 %0, %1, off offset:16\n\t"            \
            "global_load_async_to_lds_b128 %0, %1, off offset:32\n\t"            \
            "global_load_async_to_lds_b128 %0, %1, off offset:48"                \
            :: "v"(lds_), "v"(g_) : "memory");                                   \
    } while (0)

    // ---- pipeline prologue: chunk 0 into buffer 0
    ISSUE_GATHER(0, 0);

    int cb = 0;                     // buffer holding current chunk
    for (int kc = 0; kc < NCHUNK; ++kc) {
        // issue next chunk into the next rotating buffer, then wait only for
        // the current chunk's 4 async ops (in-order ASYNCcnt completion)
        if (kc + 1 < NCHUNK) {
            int nb = (cb + 1 == NBUF) ? 0 : cb + 1;
            ISSUE_GATHER(kc + 1, nb);
            if (kc + 2 < NCHUNK) {   // warm L2 two chunks ahead
                int pv = sV[svloc * NCHUNK + kc + 2];
                __builtin_prefetch(
                    ybf + (((size_t)sb * NV_ + pv) * NDIR + sdel) * NCH, 0, 3);
            }
            asm volatile("s_wait_asynccnt 0x4" ::: "memory");
        } else {
            asm volatile("s_wait_asynccnt 0x0" ::: "memory");
        }
        // raw split barrier: every thread has waited its own gather; no
        // LOADcnt/DScnt drain, so B-fragment and next-gather traffic stays live
        asm volatile("s_barrier_signal -1\n\t"
                     "s_barrier_wait -1" ::: "memory");

        // ---- compute: 2 directions x 4 N-tiles, one WMMA k-step (K=32) each
        const int arow = lane & 15;                 // A: M = lane % 16
        const int ac0  = (lane & 16) ? 16 : 0;      // A: K base per half-wave
#pragma unroll
        for (int li = 0; li < 2; ++li) {
            const int l = l0 + li;
            const __bf16* ab = &sA[cb][l * L_STRIDE + arow * ROW_STRIDE + ac0];
            bf16x8 alo = *(const bf16x8*)ab;        // K = c0 .. c0+7
            bf16x8 ahi = *(const bf16x8*)(ab + 8);  // K = c0+8 .. c0+15
            bf16x16 amat;
#pragma unroll
            for (int i = 0; i < 8; ++i) { amat[i] = alo[i]; amat[i + 8] = ahi[i]; }
#pragma unroll
            for (int nt = 0; nt < 4; ++nt) {
                const __bf16* wb =
                    wfrag + ((((size_t)(l * NCHUNK + kc) * 4 + nt) * 32 + lane) << 4);
                bf16x8 blo = *(const bf16x8*)wb;
                bf16x8 bhi = *(const bf16x8*)(wb + 8);
                bf16x16 bmat;
#pragma unroll
                for (int i = 0; i < 8; ++i) { bmat[i] = blo[i]; bmat[i + 8] = bhi[i]; }
                acc[li][nt] = __builtin_amdgcn_wmma_f32_16x16x32_bf16(
                    false, amat, false, bmat, (short)0, acc[li][nt], false, false);
            }
        }

        cb = (cb + 1 == NBUF) ? 0 : cb + 1;   // rotate; rewriter of this buffer
                                              // is 2 barriers away
    }
#undef ISSUE_GATHER

    // ---- epilogue: bias + relu + store per documented C layout
#pragma unroll
    for (int li = 0; li < 2; ++li) {
        const int l = l0 + li;
#pragma unroll
        for (int nt = 0; nt < 4; ++nt) {
#pragma unroll
            for (int g = 0; g < 8; ++g) {
                int m    = g + ((lane & 16) ? 8 : 0);   // row M
                int vloc = m >> 1, b = m & 1;           // row = vloc*2 + b
                float val = acc[li][nt][g] + bval[nt];
                val = fmaxf(val, 0.0f);
                out[(((size_t)b * NV_ + (v0 + vloc)) * NDIR + l) * NF_ +
                    nt * 16 + (lane & 15)] = val;
            }
        }
    }
}

// ---------------------------------------------------------------------------
extern "C" void kernel_launch(void* const* d_in, const int* in_sizes, int n_in,
                              void* d_out, int out_size, void* d_ws, size_t ws_size,
                              hipStream_t stream) {
    const float* y    = (const float*)d_in[0];
    const int*   ftv  = (const int*)d_in[1];
    const int*   ftd  = (const int*)d_in[2];
    const float* kern = (const float*)d_in[3];
    const float* bias = (const float*)d_in[4];
    float*       out  = (float*)d_out;

    // d_ws layout: [ybf: 30,720,000 B][wfrag: 3,145,728 B]  (~33.9 MB total)
    __bf16* ybf   = (__bf16*)d_ws;
    __bf16* wfrag = (__bf16*)((char*)d_ws + WFRAG_OFF);

    (void)in_sizes; (void)n_in; (void)out_size; (void)ws_size;

    yprep_kernel<<<(int)(YBF_ELEMS / 8 / 256), 256, 0, stream>>>(y, ybf);
    wprep_kernel<<<(16 * NCHUNK * 4 * 32 * 16) / 256, 256, 0, stream>>>(kern, wfrag);
    conv_kernel<<<NV_ / TV, 256, 0, stream>>>(ybf, ftv, ftd, bias, wfrag, out);
}
